// Atten_RNN_semiCRF_40037685133657
// MI455X (gfx1250) — matrix operand order repaired and checked by reading, not compiled, not run on hardware
//
#include <hip/hip_runtime.h>

// Semi-CRF forward scan for MI455X (gfx1250).
//
// Roofline: feats = 64 MB (2.7 us at 23.3 TB/s) and ~0.5 GFLOP -- both
// irrelevant. The binding constraint is the sequential dependence chain of
// S=32768 steps (window of 16 previous states). Design: one wave owns the
// recurrence; per step the logsumexp over (l,p) is an exp-domain
// 16x32 @ 32x32 matmul -> two V_WMMA_F32_16X16X32_F16 (A = exp(hist-base)
// resident in registers, B = exp(w1*trans)^T resident for the whole scan).
// feats is double-buffered into LDS with async LDS loads, and all span
// emission exps + the O-tag mask are baked into the chunk buffer in a bulk
// pass, so the steady-state step is branchless: 2 WMMA + 16 ds_load + FMAs
// + 2 log + 2 exp + 6 shuffles.

#define LMAX  16
#define TDIM  32
#define CHUNK 8
#define SEGS  ((CHUNK * TDIM) / (32 * 4))   // float4 segments per plane per chunk (=2)
#define NEGV  (-1.0e30f)
#define O_IDX 0

typedef __attribute__((ext_vector_type(16))) _Float16 v16h;
typedef __attribute__((ext_vector_type(8)))  float    v8f;
typedef __attribute__((ext_vector_type(4)))  int      i32x4;

#if __has_builtin(__builtin_amdgcn_global_load_async_to_lds_b128) && \
    __has_builtin(__builtin_amdgcn_s_wait_asynccnt)
#define USE_ASYNC 1
typedef __attribute__((address_space(1))) i32x4* gv4_t;   // global int4*
typedef __attribute__((address_space(3))) i32x4* lv4_t;   // LDS int4*
#else
#define USE_ASYNC 0
#endif

__device__ __forceinline__ float lse2(float a, float b) {
  float m = fmaxf(a, b);
  if (m <= -1.0e29f) return NEGV;
  return m + __logf(__expf(a - m) + __expf(b - m));
}

// Stage one chunk: plane l gets feats[l][(s0-l) .. (s0-l+CHUNK-1)][0..31]
// (16 contiguous 1 KB runs). Entries with s-l <= 0 are masked in compute,
// so clamping negative offsets to 0 is safe.
__device__ __forceinline__ void chunk_copy(const float* __restrict__ feats, int S,
                                           float* dst, int s0, int lane) {
#pragma unroll
  for (int l = 0; l < LMAX; ++l) {
    const long srcb = ((long)l * S + (long)(s0 - l)) * TDIM;
    float* dpl = dst + l * (CHUNK * TDIM);
#pragma unroll
    for (int seg = 0; seg < SEGS; ++seg) {
      long e = srcb + (long)(seg * 32 + lane) * 4;
      if (e < 0) e = 0;
      float* d = dpl + (seg * 32 + lane) * 4;
#if USE_ASYNC
      __builtin_amdgcn_global_load_async_to_lds_b128(
          (gv4_t)(i32x4*)(feats + e), (lv4_t)(i32x4*)d, 0, 0);
#else
      const float4 v = *(const float4*)(feats + e);
      *(float4*)d = v;
#endif
    }
  }
}

// One recurrence step. BOUND=true only for s < LMAX (and tails): adds the
// span-fits mask and the sentence-start boundary candidate. All masking is
// branchless (multiply by 0/1, cndmask) to keep the chain short.
template <bool BOUND>
__device__ __forceinline__ void do_step(
    int s, int i, int lane, int nloc, bool hi,
    const float* __restrict__ Wbuf,     // transformed chunk: exp(w2*fd), O-mask baked
    const float* __restrict__ bemit,    // raw boundary emissions [16][32]
    _Float16* s_new,                    // 32-entry staging row
    v16h& af, const v16h& bf0, const v16h& bf1,
    float tb0_a, float tb0_b, float w2, float& base, int& head) {
  // G[l_phys][t] = sum_p exp(hist[l_phys][p]-base) * exp(w1*trans[t][p])
  v8f c0 = {}, c1 = {};
  c0 = __builtin_amdgcn_wmma_f32_16x16x32_f16(false, af, false, bf0, (short)0, c0, false, false);
  c1 = __builtin_amdgcn_wmma_f32_16x16x32_f16(false, af, false, bf1, (short)0, c1, false, false);

  // D layout: VGPR r -> row M = r + 8*hi, column N = lane&15.
  float acc0 = 0.f, acc1 = 0.f;
#pragma unroll
  for (int r = 0; r < 8; ++r) {
    const int l_phys = r + (hi ? 8 : 0);
    const int l_log  = (l_phys - head) & 15;
    const float* row = Wbuf + (l_log * CHUNK + i) * TDIM;
    float w_0 = row[nloc];
    float w_1 = row[nloc + 16];
    if (BOUND) {
      const float m = ((l_log + 1) <= s) ? 1.f : 0.f;   // span fits in prefix
      w_0 *= m;
      w_1 *= m;
    }
    acc0 = fmaf(w_0, c0[r], acc0);
    acc1 = fmaf(w_1, c1[r], acc1);
  }
  acc0 += __shfl_xor(acc0, 16, 32);   // fold the two M-halves per tag column
  acc1 += __shfl_xor(acc1, 16, 32);

  const float l0 = (acc0 > 0.f) ? (base + __logf(acc0)) : NEGV;
  const float l1 = (acc1 > 0.f) ? (base + __logf(acc1)) : NEGV;
  float fv0, fv1;
  if (BOUND) {
    const bool  bok = (s < LMAX);
    const float bs0 = (bok && (nloc != O_IDX || s == 0))
                      ? (tb0_a + w2 * bemit[(s & (LMAX - 1)) * TDIM + nloc]) : NEGV;
    const float bs1 = bok ? (tb0_b + w2 * bemit[(s & (LMAX - 1)) * TDIM + nloc + 16]) : NEGV;
    fv0 = lse2(l0, bs0);
    fv1 = lse2(l1, bs1);
  } else {
    fv0 = l0;
    fv1 = l1;
  }

  // new shift = max over all tags
  float mx = fmaxf(fv0, fv1);
#pragma unroll
  for (int d = 1; d < 16; d <<= 1) mx = fmaxf(mx, __shfl_xor(mx, d, 32));

  // rotate head, publish the new row, rescale + select into resident A.
  const float scale = __expf(base - mx);
  head = (head + 15) & 15;
  s_new[lane] = (_Float16)__expf((hi ? fv1 : fv0) - mx);  // lane == tag index

  v16h cand;
  const _Float16* p0 = s_new + (hi ? 8 : 0);
#pragma unroll
  for (int j = 0; j < 8; ++j) {
    const int k0 = ((j < 4) ? 0 : 16) + 2 * (j & 3);      // A layout K pattern
    cand[2*j]   = p0[k0];
    cand[2*j+1] = p0[k0 + 1];
  }
  const v16h scaled = af * (_Float16)scale;
  af = (nloc == head) ? cand : scaled;   // only the 2 lanes owning row `head` take cand
  base = mx;
}

__global__ __launch_bounds__(32, 1)
void semicrf_fwd(const float* __restrict__ feats,
                 const float* __restrict__ trans,
                 const float* __restrict__ tbound,
                 const float* __restrict__ w1p,
                 const float* __restrict__ w2p,
                 float* __restrict__ out, int S) {
  __shared__ __align__(16) float    s_buf[2][LMAX][CHUNK][TDIM];  // 32 KB double-buffered
  __shared__ __align__(16) float    s_bemit[LMAX][TDIM];          // raw boundary emissions
  __shared__ __align__(16) _Float16 s_new[TDIM];                  // newest-row staging

  const int  lane = threadIdx.x;     // 0..31 (wave32)
  const int  nloc = lane & 15;
  const bool hi   = lane >= 16;
  const float w1 = w1p[0], w2 = w2p[0];

  // ---- B fragments (constant for whole scan): B[k][t] = exp(w1*trans[t][k]).
  v16h bf0, bf1;
#pragma unroll
  for (int j = 0; j < 8; ++j) {
    const int kb = ((j < 4) ? 0 : 16) + (hi ? 8 : 0) + 2 * (j & 3);
    bf0[2*j]   = (_Float16)__expf(w1 * trans[nloc * TDIM + kb]);
    bf0[2*j+1] = (_Float16)__expf(w1 * trans[nloc * TDIM + kb + 1]);
    bf1[2*j]   = (_Float16)__expf(w1 * trans[(nloc + 16) * TDIM + kb]);
    bf1[2*j+1] = (_Float16)__expf(w1 * trans[(nloc + 16) * TDIM + kb + 1]);
  }

  const float tb0_a = w1 * tbound[nloc];
  const float tb0_b = w1 * tbound[nloc + 16];
  const float tb1_a = w1 * tbound[TDIM + nloc];
  const float tb1_b = w1 * tbound[TDIM + nloc + 16];

  // boundary emissions: feats[s, 0, :] for s < LMAX
#pragma unroll
  for (int r = 0; r < LMAX; ++r)
    s_bemit[r][lane] = feats[(long)r * S * TDIM + lane];

  v16h af = {};        // resident A fragment: empty history -> exp(NEG) == 0
  float base = NEGV;   // running log-domain shift
  int head = 0;        // physical slot of logical l=0
  int buf = 0;

  chunk_copy(feats, S, &s_buf[0][0][0][0], 0, lane);

  for (int s0 = 0; s0 < S; s0 += CHUNK) {
    const int next = s0 + CHUNK;
    if (next < S) {
      chunk_copy(feats, S, &s_buf[buf ^ 1][0][0][0], next, lane);
#if USE_ASYNC
      __builtin_amdgcn_s_wait_asynccnt(LMAX * SEGS);  // current chunk done; next stays in flight
#endif
    } else {
#if USE_ASYNC
      __builtin_amdgcn_s_wait_asynccnt(0);
#endif
    }

    // ---- bulk transform (off the critical chain): W = exp(w2*fd), and bake
    // the O-tag mask (t==O_IDX only allows span length 1, i.e. l==0).
    float* W = &s_buf[buf][0][0][0];
#pragma unroll 4
    for (int it = 0; it < (LMAX * CHUNK * TDIM) / (32 * 4); ++it) {  // 32 iters
      const int idx = (it * 32 + lane) * 4;
      float4 v = *(float4*)(W + idx);
      const int l = idx >> 8;                       // CHUNK*TDIM floats per plane
      v.x = (((idx & 31) == O_IDX) && l != 0) ? 0.f : __expf(w2 * v.x);
      v.y = __expf(w2 * v.y);
      v.z = __expf(w2 * v.z);
      v.w = __expf(w2 * v.w);
      *(float4*)(W + idx) = v;
    }

    if (s0 >= LMAX && next <= S) {
      // steady state: no masks, fully unrolled, branchless
#pragma unroll
      for (int i = 0; i < CHUNK; ++i)
        do_step<false>(s0 + i, i, lane, nloc, hi, W, &s_bemit[0][0], s_new,
                       af, bf0, bf1, tb0_a, tb0_b, w2, base, head);
    } else {
      const int ilim = (S - s0 < CHUNK) ? (S - s0) : CHUNK;
      for (int i = 0; i < ilim; ++i)
        do_step<true>(s0 + i, i, lane, nloc, hi, W, &s_bemit[0][0], s_new,
                      af, bf0, bf1, tb0_a, tb0_b, w2, base, head);
    }
    buf ^= 1;
  }

  // terminal: s_new holds exp(hist[0][t] - base) after the last step.
  float tot = (float)s_new[nloc]      * __expf(tb1_a)
            + (float)s_new[nloc + 16] * __expf(tb1_b);
#pragma unroll
  for (int d = 1; d < 16; d <<= 1) tot += __shfl_xor(tot, d, 32);
  if (lane == 0) out[0] = base + __logf(tot);
}

extern "C" void kernel_launch(void* const* d_in, const int* in_sizes, int n_in,
                              void* d_out, int out_size, void* d_ws, size_t ws_size,
                              hipStream_t stream) {
  (void)n_in; (void)out_size; (void)d_ws; (void)ws_size;
  const float* feats  = (const float*)d_in[0];
  const float* trans  = (const float*)d_in[1];
  const float* tbound = (const float*)d_in[2];
  const float* w1     = (const float*)d_in[3];
  const float* w2     = (const float*)d_in[4];
  const int S = in_sizes[0] / (LMAX * TDIM);   // feats is [L, S, T]
  hipLaunchKernelGGL(semicrf_fwd, dim3(1), dim3(32), 0, stream,
                     feats, trans, tbound, w1, w2, (float*)d_out, S);
}
